// Pixel_SSCM_68006512165285
// MI455X (gfx1250) — matrix operand order repaired and checked by reading, not compiled, not run on hardware
//
#include <hip/hip_runtime.h>
#include <hip/hip_bf16.h>

// ---------------------------------------------------------------------------
// Pixel cross-batch attention block on gfx1250 (MI455X), wave32 + WMMA f16/f32
// ---------------------------------------------------------------------------

typedef __attribute__((ext_vector_type(16))) _Float16 v16h;
typedef __attribute__((ext_vector_type(8)))  _Float16 v8h;
typedef __attribute__((ext_vector_type(8)))  float    v8f;

#define T_TOK 2048   // B * N  (8 * 256)
#define NTOK  256    // tokens per batch (GRID*GRID)
#define CIN   256
#define EMB   512
#define NH    8
#define HD    64
#define LN_EPS 1e-5f

// ---- WMMA fragment helpers (wave32, v_wmma_f32_16x16x32_f16) ---------------

// A fragment: 16x32 tile of row-major A (lda in halves). k0 multiple of 32.
__device__ __forceinline__ v16h load_a_frag(const _Float16* A, int lda, int row0, int k0) {
  int lane = threadIdx.x & 31;
  int hi16 = lane >> 4;                         // 0: K 0..15, 1: K 16..31
  const _Float16* base = A + (size_t)(row0 + (lane & 15)) * lda + k0 + hi16 * 16;
  v8h lo = *(const v8h*)(base);                 // K = hi16*16 + 0..7
  v8h hi = *(const v8h*)(base + 8);             // K = hi16*16 + 8..15
  v16h r;
#pragma unroll
  for (int i = 0; i < 8; ++i) { r[i] = lo[i]; r[i + 8] = hi[i]; }
  return r;
}

// B fragment: 32x16 tile where Bn is stored N-major: Bn[n][k], ldb in halves.
// Lane n holds K = (lane>>4)*16 + 0..15 -> one contiguous 32-byte load.
__device__ __forceinline__ v16h load_b_frag(const _Float16* Bn, int ldb, int col0, int k0) {
  int lane = threadIdx.x & 31;
  const _Float16* base = Bn + (size_t)(col0 + (lane & 15)) * ldb + k0 + (lane >> 4) * 16;
  return *(const v16h*)(base);
}

__device__ __forceinline__ v8f wmma16(v16h a, v16h b, v8f c) {
  // (neg_a, A, neg_b, B, c_mod, C, reuse_a, reuse_b)
  return __builtin_amdgcn_wmma_f32_16x16x32_f16(false, a, false, b, (short)0, c,
                                                false, false);
}

// ---- Stage 0: layout conversions ------------------------------------------

// x [B][Cin][N] f32  ->  Xt [T][Cin] f16   (t = b*256 + hw)
__global__ void k_tokens(const float* __restrict__ x, _Float16* __restrict__ Xt) {
  int idx = blockIdx.x * blockDim.x + threadIdx.x;
  if (idx >= T_TOK * CIN) return;
  int t = idx >> 8;            // idx = t*256 + i
  int i = idx & 255;
  int b = t >> 8, hw = t & 255;
  Xt[idx] = (_Float16)x[(size_t)b * CIN * NTOK + (size_t)i * NTOK + hw];
}

__global__ void k_h16(const float* __restrict__ w, _Float16* __restrict__ o, int n) {
  int i = blockIdx.x * blockDim.x + threadIdx.x;
  if (i < n) o[i] = (_Float16)w[i];
}

// ---- Stage 1: fused QKV projection GEMM  [2048x256]@[256x512] --------------
// mode 0/1: out (q or k) -> qk[head][t][c]   (head = o&7, c = o>>3)
// mode 2:   out (v)      -> vT[head][c][t]
__global__ void k_qkv(const _Float16* __restrict__ Xt, const _Float16* __restrict__ W16,
                      const float* __restrict__ bias, _Float16* __restrict__ outQK,
                      _Float16* __restrict__ outVT, int mode) {
  int wave = threadIdx.x >> 5, lane = threadIdx.x & 31;
  int row0 = blockIdx.x * 16;          // 128 M strips
  int col0 = wave * 64;                // 8 waves x 64 cols = 512
  v8f acc[4] = {};
  for (int k0 = 0; k0 < CIN; k0 += 32) {
    v16h a = load_a_frag(Xt, CIN, row0, k0);
#pragma unroll
    for (int j = 0; j < 4; ++j) {
      v16h b = load_b_frag(W16, CIN, col0 + j * 16, k0);
      acc[j] = wmma16(a, b, acc[j]);
    }
  }
  int n_lane = lane & 15;
  int m_base = row0 + (lane >> 4) * 8;
#pragma unroll
  for (int j = 0; j < 4; ++j) {
    int o = col0 + j * 16 + n_lane;
    float bo = bias[o];
    int head = o & 7, c = o >> 3;
#pragma unroll
    for (int v = 0; v < 8; ++v) {
      int t = m_base + v;
      float val = acc[j][v] + bo;
      if (mode == 2)
        outVT[((size_t)head * HD + c) * T_TOK + t] = (_Float16)val;
      else
        outQK[((size_t)head * T_TOK + t) * HD + c] = (_Float16)val;
    }
  }
}

// ---- Stage 2a: logits S = Q_h @ K_h^T  [2048x64]@[64x2048] -----------------
__global__ void k_logits(const _Float16* __restrict__ qh, const _Float16* __restrict__ kh,
                         float* __restrict__ S, int h) {
  const _Float16* Q = qh + (size_t)h * T_TOK * HD;
  const _Float16* K = kh + (size_t)h * T_TOK * HD;   // [m][c] = N-major for B frags
  int wave = threadIdx.x >> 5, lane = threadIdx.x & 31;
  int row0 = blockIdx.x * 16;
  int col0 = blockIdx.y * 512 + wave * 64;
  v8f acc[4] = {};
#pragma unroll
  for (int k0 = 0; k0 < HD; k0 += 32) {
    v16h a = load_a_frag(Q, HD, row0, k0);
#pragma unroll
    for (int j = 0; j < 4; ++j) {
      v16h b = load_b_frag(K, HD, col0 + j * 16, k0);
      acc[j] = wmma16(a, b, acc[j]);
    }
  }
  int n_lane = lane & 15;
  int m_base = row0 + (lane >> 4) * 8;
#pragma unroll
  for (int j = 0; j < 4; ++j) {
    int n = col0 + j * 16 + n_lane;
#pragma unroll
    for (int v = 0; v < 8; ++v)
      S[(size_t)(m_base + v) * T_TOK + n] = acc[j][v];
  }
}

// ---- Stage 2b: chunked softmax (per row, per 256-wide q-chunk) -> f16 P ----
__global__ void k_softmax(const float* __restrict__ S, _Float16* __restrict__ P) {
  __shared__ float red[256];
  int tid = threadIdx.x;
  size_t base = (size_t)blockIdx.x * T_TOK + (size_t)blockIdx.y * 256;
  float x = S[base + tid];
  red[tid] = x; __syncthreads();
  for (int s = 128; s > 0; s >>= 1) {
    if (tid < s) red[tid] = fmaxf(red[tid], red[tid + s]);
    __syncthreads();
  }
  float mx = red[0]; __syncthreads();
  float e = __expf(x - mx);
  red[tid] = e; __syncthreads();
  for (int s = 128; s > 0; s >>= 1) {
    if (tid < s) red[tid] += red[tid + s];
    __syncthreads();
  }
  float inv = 1.0f / red[0];
  P[base + tid] = (_Float16)(e * inv);
}

// ---- Stage 2c: O_h = P_h @ V_h  [2048x2048]@[2048x64] ----------------------
// V stored transposed (vT[h][c][t]) so B fragments are contiguous.
// Scatter into O16[t][o] f16 with o = c*8 + h.
__global__ void k_pv(const _Float16* __restrict__ P, const _Float16* __restrict__ vhT,
                     _Float16* __restrict__ O16, int h) {
  const _Float16* Vt = vhT + (size_t)h * HD * T_TOK;
  int wave = threadIdx.x >> 5, lane = threadIdx.x & 31;
  int row0 = blockIdx.x * 16;
  int col0 = wave * 16;                // 4 waves cover 64 cols
  v8f acc = {};
  for (int k0 = 0; k0 < T_TOK; k0 += 32) {
    if (k0 + 128 < T_TOK)              // gfx1250 global_prefetch for the long K loop
      __builtin_prefetch(P + (size_t)(row0 + (threadIdx.x & 15)) * T_TOK + k0 + 128, 0, 1);
    v16h a = load_a_frag(P, T_TOK, row0, k0);
    v16h b = load_b_frag(Vt, T_TOK, col0, k0);
    acc = wmma16(a, b, acc);
  }
  int n_lane = lane & 15;
  int m_base = row0 + (lane >> 4) * 8;
  int c = col0 + n_lane;
  int o = c * NH + h;
#pragma unroll
  for (int v = 0; v < 8; ++v)
    O16[(size_t)(m_base + v) * EMB + o] = (_Float16)acc[v];
}

// ---- Stage 3: a = O @ Wa^T + ba + x1  (f32, for LN precision) --------------
__global__ void k_aproj(const _Float16* __restrict__ O16, const _Float16* __restrict__ Wa16,
                        const float* __restrict__ ba, const float* __restrict__ x1,
                        float* __restrict__ Abuf) {
  int wave = threadIdx.x >> 5, lane = threadIdx.x & 31;
  int row0 = blockIdx.x * 16;
  v8f acc[2] = {};
  for (int k0 = 0; k0 < EMB; k0 += 32) {
    v16h a = load_a_frag(O16, EMB, row0, k0);
#pragma unroll
    for (int j = 0; j < 2; ++j) {
      v16h b = load_b_frag(Wa16, EMB, (wave * 2 + j) * 16, k0);
      acc[j] = wmma16(a, b, acc[j]);
    }
  }
  int n_lane = lane & 15;
  int m_base = row0 + (lane >> 4) * 8;
#pragma unroll
  for (int j = 0; j < 2; ++j) {
    int co = (wave * 2 + j) * 16 + n_lane;
    float bc = ba[co];
#pragma unroll
    for (int v = 0; v < 8; ++v) {
      int t = m_base + v;
      int b = t >> 8, hw = t & 255;
      Abuf[(size_t)t * CIN + co] =
          acc[j][v] + bc + x1[(size_t)b * CIN * NTOK + (size_t)co * NTOK + hw];
    }
  }
}

// ---- Stage 4: LayerNorm over channels -> f16 -------------------------------
__global__ void k_ln(const float* __restrict__ Abuf, const float* __restrict__ g,
                     const float* __restrict__ bta, _Float16* __restrict__ Tn) {
  __shared__ float red[256];
  int t = blockIdx.x, i = threadIdx.x;
  float x = Abuf[(size_t)t * CIN + i];
  red[i] = x; __syncthreads();
  for (int s = 128; s > 0; s >>= 1) {
    if (i < s) red[i] += red[i + s];
    __syncthreads();
  }
  float mu = red[0] * (1.0f / 256.0f); __syncthreads();
  float d = x - mu;
  red[i] = d * d; __syncthreads();
  for (int s = 128; s > 0; s >>= 1) {
    if (i < s) red[i] += red[i + s];
    __syncthreads();
  }
  float var = red[0] * (1.0f / 256.0f);
  float y = d * rsqrtf(var + LN_EPS) * g[i] + bta[i];
  Tn[(size_t)t * CIN + i] = (_Float16)y;
}

// ---- Stage 5: y = relu(Tn @ Wl^T + bl), scatter to [b][co][hw] f32 ---------
__global__ void k_final(const _Float16* __restrict__ Tn, const _Float16* __restrict__ Wl16,
                        const float* __restrict__ bl, float* __restrict__ out) {
  int wave = threadIdx.x >> 5, lane = threadIdx.x & 31;
  int row0 = blockIdx.x * 16;
  v8f acc[2] = {};
  for (int k0 = 0; k0 < CIN; k0 += 32) {
    v16h a = load_a_frag(Tn, CIN, row0, k0);
#pragma unroll
    for (int j = 0; j < 2; ++j) {
      v16h b = load_b_frag(Wl16, CIN, (wave * 2 + j) * 16, k0);
      acc[j] = wmma16(a, b, acc[j]);
    }
  }
  int n_lane = lane & 15;
  int m_base = row0 + (lane >> 4) * 8;
#pragma unroll
  for (int j = 0; j < 2; ++j) {
    int co = (wave * 2 + j) * 16 + n_lane;
    float bc = bl[co];
#pragma unroll
    for (int v = 0; v < 8; ++v) {
      int t = m_base + v;
      float y = acc[j][v] + bc;
      y = y > 0.0f ? y : 0.0f;
      out[(size_t)(t >> 8) * CIN * NTOK + (size_t)co * NTOK + (t & 255)] = y;
    }
  }
}

// ---------------------------------------------------------------------------

extern "C" void kernel_launch(void* const* d_in, const int* in_sizes, int n_in,
                              void* d_out, int out_size, void* d_ws, size_t ws_size,
                              hipStream_t stream) {
  (void)in_sizes; (void)n_in; (void)out_size; (void)ws_size;
  const float* x1   = (const float*)d_in[0];
  const float* x2   = (const float*)d_in[1];
  const float* Wq   = (const float*)d_in[2];
  const float* bq   = (const float*)d_in[3];
  const float* Wk   = (const float*)d_in[4];
  const float* bk   = (const float*)d_in[5];
  const float* Wv   = (const float*)d_in[6];
  const float* bv   = (const float*)d_in[7];
  const float* Wa   = (const float*)d_in[8];
  const float* ba   = (const float*)d_in[9];
  const float* ln_g = (const float*)d_in[10];
  const float* ln_b = (const float*)d_in[11];
  const float* Wl   = (const float*)d_in[12];
  const float* bl   = (const float*)d_in[13];

  size_t off = 0;
  auto wsalloc = [&](size_t bytes) -> void* {
    void* p = (char*)d_ws + off;
    off += (bytes + 255) & ~(size_t)255;
    return p;
  };
  _Float16* Xt1  = (_Float16*)wsalloc((size_t)T_TOK * CIN * 2);
  _Float16* Xt2  = (_Float16*)wsalloc((size_t)T_TOK * CIN * 2);
  _Float16* Wq16 = (_Float16*)wsalloc((size_t)EMB * CIN * 2);
  _Float16* Wk16 = (_Float16*)wsalloc((size_t)EMB * CIN * 2);
  _Float16* Wv16 = (_Float16*)wsalloc((size_t)EMB * CIN * 2);
  _Float16* Wa16 = (_Float16*)wsalloc((size_t)CIN * EMB * 2);
  _Float16* Wl16 = (_Float16*)wsalloc((size_t)CIN * CIN * 2);
  _Float16* qh   = (_Float16*)wsalloc((size_t)NH * T_TOK * HD * 2);
  _Float16* kh   = (_Float16*)wsalloc((size_t)NH * T_TOK * HD * 2);
  _Float16* vhT  = (_Float16*)wsalloc((size_t)NH * HD * T_TOK * 2);
  _Float16* O16  = (_Float16*)wsalloc((size_t)T_TOK * EMB * 2);
  float*    Abuf = (float*)   wsalloc((size_t)T_TOK * CIN * 4);
  _Float16* Tn16 = (_Float16*)wsalloc((size_t)T_TOK * CIN * 2);
  float*    S    = (float*)   wsalloc((size_t)T_TOK * T_TOK * 4);   // per-head, reused
  _Float16* P    = (_Float16*)wsalloc((size_t)T_TOK * T_TOK * 2);   // per-head, reused

  // Stage 0: conversions
  int nXt = T_TOK * CIN;
  k_tokens<<<(nXt + 255) / 256, 256, 0, stream>>>(x1, Xt1);
  k_tokens<<<(nXt + 255) / 256, 256, 0, stream>>>(x2, Xt2);
  k_h16<<<(EMB * CIN + 255) / 256, 256, 0, stream>>>(Wq, Wq16, EMB * CIN);
  k_h16<<<(EMB * CIN + 255) / 256, 256, 0, stream>>>(Wk, Wk16, EMB * CIN);
  k_h16<<<(EMB * CIN + 255) / 256, 256, 0, stream>>>(Wv, Wv16, EMB * CIN);
  k_h16<<<(CIN * EMB + 255) / 256, 256, 0, stream>>>(Wa, Wa16, CIN * EMB);
  k_h16<<<(CIN * CIN + 255) / 256, 256, 0, stream>>>(Wl, Wl16, CIN * CIN);

  // Stage 1: Q/K/V projections (WMMA GEMMs)
  k_qkv<<<128, 256, 0, stream>>>(Xt1, Wq16, bq, qh, (_Float16*)nullptr, 0);
  k_qkv<<<128, 256, 0, stream>>>(Xt2, Wk16, bk, kh, (_Float16*)nullptr, 1);
  k_qkv<<<128, 256, 0, stream>>>(Xt2, Wv16, bv, (_Float16*)nullptr, vhT, 2);

  // Stage 2: per-head attention, S/P buffers reused (stream-ordered)
  for (int h = 0; h < NH; ++h) {
    k_logits<<<dim3(128, 4), 256, 0, stream>>>(qh, kh, S, h);
    k_softmax<<<dim3(T_TOK, 8), 256, 0, stream>>>(S, P);
    k_pv<<<128, 128, 0, stream>>>(P, vhT, O16, h);
  }

  // Stage 3..5: Wa projection + residual, LayerNorm, final linear + ReLU
  k_aproj<<<128, 256, 0, stream>>>(O16, Wa16, ba, x1, Abuf);
  k_ln<<<T_TOK, 256, 0, stream>>>(Abuf, ln_g, ln_b, Tn16);
  k_final<<<128, 256, 0, stream>>>(Tn16, Wl16, bl, (float*)d_out);
}
